// ContextualActorCritic_81260781240456
// MI455X (gfx1250) — compile-verified, matrix operand order checked by reading, not run on hardware
//
#include <hip/hip_runtime.h>

// ---------------------------------------------------------------------------
// Types
// ---------------------------------------------------------------------------
typedef __attribute__((ext_vector_type(16))) _Float16 v16h;
typedef __attribute__((ext_vector_type(8)))  _Float16 v8h;
typedef __attribute__((ext_vector_type(4)))  _Float16 v4h;
typedef __attribute__((ext_vector_type(8)))  float    v8f;
typedef __attribute__((ext_vector_type(4)))  int      vi4;

union FragU { vi4 q[2]; v16h h; };
union H8U   { vi4 q;    v8h  h; };

#define BATCH   65536
#define OBS     64
#define HDIM    256
#define NLEAF   16
#define AMAX    32
#define XPAD    72      // 64 + 8 halves  -> 144B row stride (conflict-free b128)
#define HPAD    264     // 256 + 8 halves -> 528B row stride (conflict-free b128)
#define WAVES   8
#define ROWS_PER_WAVE 16
#define ROWS_PER_BLOCK (WAVES * ROWS_PER_WAVE)   // 128

// f16 weight workspace layout (offsets in halves)
#define OFF_W1T 0                         // [256][64]
#define OFF_W2T 16384                     // [256][256]
#define OFF_V1T 81920                     // [256][64]
#define OFF_V2T 98304                     // [256][256]
#define OFF_WH  163840                    // [16][32][256] (k-contiguous, no transpose)
#define WS_HALVES 294912                  // 576 KB

// ---------------------------------------------------------------------------
// Fast math helpers (map to v_exp_f32 / v_log_f32 / v_rcp_f32)
// ---------------------------------------------------------------------------
__device__ __forceinline__ float fast_tanh(float x) {
  x = fminf(10.0f, fmaxf(-10.0f, x));
  float t = __builtin_amdgcn_exp2f(x * 2.8853900817779268f);  // exp(2x)
  return (t - 1.0f) * __builtin_amdgcn_rcpf(t + 1.0f);
}

// ---------------------------------------------------------------------------
// Fragment loaders.
// B fragment (16-bit, 32x16): lane L<16 -> col N=L,    K = {kc..kc+7, kc+16..kc+23}
//                             lane L>=16 -> col N=L-16, K = {kc+8..kc+15, kc+24..kc+31}
// Weights are pre-stored as WT[n][k] (k contiguous, stride kStride halves).
// A fragment (16-bit, 16x32): identical per-lane gather from a row-major
// activation tile in LDS (row = lane%16).
// ---------------------------------------------------------------------------
__device__ __forceinline__ v16h ldg_frag(const _Float16* __restrict__ w,
                                         int kStride, int nBase, int kc,
                                         int lh, int hi) {
  const _Float16* p = w + (size_t)(nBase + lh) * kStride + (kc + hi * 8);
  FragU u;
  u.q[0] = *(const vi4*)(p);
  u.q[1] = *(const vi4*)(p + 16);
  return u.h;
}

__device__ __forceinline__ v16h lds_frag(const _Float16* s, int rStride,
                                         int kc, int lh, int hi) {
  const _Float16* p = s + lh * rStride + (kc + hi * 8);
  FragU u;
  u.q[0] = *(const vi4*)(p);
  u.q[1] = *(const vi4*)(p + 16);
  return u.h;
}

__device__ __forceinline__ v8f wmma16(v16h a, v16h b, v8f c) {
  return __builtin_amdgcn_wmma_f32_16x16x32_f16(false, a, false, b,
                                                (short)0, c, false, false);
}

// ---------------------------------------------------------------------------
// Kernel 0: convert/transpose f32 weights -> f16 fragment-friendly layout
// ---------------------------------------------------------------------------
__global__ void cac_convert_weights(const float* __restrict__ W1,
                                    const float* __restrict__ W2,
                                    const float* __restrict__ Wh,
                                    const float* __restrict__ V1,
                                    const float* __restrict__ V2,
                                    _Float16* __restrict__ ws) {
  for (int i = blockIdx.x * blockDim.x + threadIdx.x; i < WS_HALVES;
       i += gridDim.x * blockDim.x) {
    float v;
    if (i < OFF_W2T) {                 // W1T[n][k], K=64
      int j = i - OFF_W1T; int n = j >> 6; int k = j & 63;
      v = W1[k * HDIM + n];
    } else if (i < OFF_V1T) {          // W2T[n][k], K=256
      int j = i - OFF_W2T; int n = j >> 8; int k = j & 255;
      v = W2[k * HDIM + n];
    } else if (i < OFF_V2T) {          // V1T[n][k], K=64
      int j = i - OFF_V1T; int n = j >> 6; int k = j & 63;
      v = V1[k * HDIM + n];
    } else if (i < OFF_WH) {           // V2T[n][k], K=256
      int j = i - OFF_V2T; int n = j >> 8; int k = j & 255;
      v = V2[k * HDIM + n];
    } else {                           // Wh straight copy ([l][a][k])
      v = Wh[i - OFF_WH];
    }
    ws[i] = (_Float16)v;
  }
}

// ---------------------------------------------------------------------------
// Kernel 1: fully fused actor-critic forward. One wave owns 16 rows, no
// inter-wave communication -> no barriers; LDS used only for the C->A
// cross-lane layout shuffle (intra-wave, DScnt-ordered).
//
// Single activation buffer `h`: dense256 hoists all 8 A-fragments into
// registers before any write-back, and the LDS pipe is in-order per wave,
// so the layer output can legally overwrite its own input. 12.5 KB/wave,
// 100 KB/workgroup -> 3 workgroups (24 waves) resident per 320 KB WGP.
// ---------------------------------------------------------------------------
struct __align__(16) WaveSmem {
  _Float16 x[ROWS_PER_WAVE * XPAD];      // 2304 B (alive until value layer 1)
  _Float16 h[ROWS_PER_WAVE * HPAD];      // 8448 B (in-place layer chain)
  float    logits[ROWS_PER_WAVE * AMAX]; // 2048 B
};                                        // 12800 B * 8 waves = 100 KB

__global__ __launch_bounds__(256)
void cac_fused(const float* __restrict__ obs,
               const int*   __restrict__ actions,
               const int*   __restrict__ leaf_ids,
               const int*   __restrict__ counts,
               const float* __restrict__ b1,  const float* __restrict__ b2,
               const float* __restrict__ bh,
               const float* __restrict__ bv1, const float* __restrict__ bv2,
               const float* __restrict__ V3,  const float* __restrict__ bv3,
               const _Float16* __restrict__ ws,
               float* __restrict__ out_logp,
               float* __restrict__ out_ent,
               float* __restrict__ out_val) {
  __shared__ WaveSmem smem[WAVES];

  const int wave = threadIdx.x >> 5;
  const int lane = threadIdx.x & 31;
  const int lh   = lane & 15;     // column / row-within-16
  const int hi   = lane >> 4;     // half-wave select
  const int rowBase = blockIdx.x * ROWS_PER_BLOCK + wave * ROWS_PER_WAVE;

  WaveSmem& sm = smem[wave];
  const _Float16* w1t = ws + OFF_W1T;
  const _Float16* w2t = ws + OFF_W2T;
  const _Float16* v1t = ws + OFF_V1T;
  const _Float16* v2t = ws + OFF_V2T;
  const _Float16* wht = ws + OFF_WH;

  // ---- Stage X: load this wave's 16 obs rows into LDS as f16 -------------
  {
    const int r    = lane >> 1;           // 2 lanes per row
    const int kOff = (lane & 1) * 32;     // each lane: 32 contiguous floats
    const float* src = obs + (size_t)(rowBase + r) * OBS + kOff;
    _Float16* dst = sm.x + r * XPAD + kOff;
#pragma unroll
    for (int c = 0; c < 32; c += 4) {
      float4 v = *(const float4*)(src + c);
      v4h h = { (_Float16)v.x, (_Float16)v.y, (_Float16)v.z, (_Float16)v.w };
      *(v4h*)(dst + c) = h;
    }
  }

  // ---- dense64: x(LDS) @ WT(64-deep) + bias -> tanh -> h(LDS) ------------
  auto dense64 = [&](const _Float16* wt, const float* bias) {
    v16h a0 = lds_frag(sm.x, XPAD, 0, lh, hi);
    v16h a1 = lds_frag(sm.x, XPAD, 32, lh, hi);
    for (int nt = 0; nt < 16; ++nt) {
      v8f c = (v8f)0.0f;
      c = wmma16(a0, ldg_frag(wt, OBS, nt * 16, 0, lh, hi), c);
      c = wmma16(a1, ldg_frag(wt, OBS, nt * 16, 32, lh, hi), c);
      float bv = bias[nt * 16 + lh];
#pragma unroll
      for (int r = 0; r < 8; ++r)
        sm.h[(r + 8 * hi) * HPAD + nt * 16 + lh] = (_Float16)fast_tanh(c[r] + bv);
    }
  };

  // ---- dense256: h(LDS) @ WT(256-deep) + bias -> tanh -> h(LDS, in place) -
  auto dense256 = [&](const _Float16* wt, const float* bias) {
    v16h a[8];
#pragma unroll
    for (int ks = 0; ks < 8; ++ks) a[ks] = lds_frag(sm.h, HPAD, ks * 32, lh, hi);
    for (int nt = 0; nt < 16; ++nt) {
      v8f c = (v8f)0.0f;
#pragma unroll
      for (int ks = 0; ks < 8; ++ks)
        c = wmma16(a[ks], ldg_frag(wt, HDIM, nt * 16, ks * 32, lh, hi), c);
      float bv = bias[nt * 16 + lh];
#pragma unroll
      for (int r = 0; r < 8; ++r)
        sm.h[(r + 8 * hi) * HPAD + nt * 16 + lh] = (_Float16)fast_tanh(c[r] + bv);
    }
  };

  // ---- Policy backbone ----------------------------------------------------
  dense64(w1t, b1);        // H1 = tanh(X W1 + b1)
  dense256(w2t, b2);       // H  = tanh(H1 W2 + b2), in place

  // ---- Per-leaf heads: compute every leaf's logits, keep own leaf only ----
  {
    v16h a[8];               // capture H into registers; LDS h is free after
#pragma unroll
    for (int ks = 0; ks < 8; ++ks) a[ks] = lds_frag(sm.h, HPAD, ks * 32, lh, hi);
    int myleaf[8];
#pragma unroll
    for (int r = 0; r < 8; ++r)
      myleaf[r] = leaf_ids[rowBase + r + 8 * hi];

    for (int l = 0; l < NLEAF; ++l) {
      const _Float16* wl = wht + (size_t)l * AMAX * HDIM;
      for (int nt = 0; nt < 2; ++nt) {
        v8f c = (v8f)0.0f;
#pragma unroll
        for (int ks = 0; ks < 8; ++ks)
          c = wmma16(a[ks], ldg_frag(wl, HDIM, nt * 16, ks * 32, lh, hi), c);
        float bv = bh[l * AMAX + nt * 16 + lh];
#pragma unroll
        for (int r = 0; r < 8; ++r)
          if (myleaf[r] == l)
            sm.logits[(r + 8 * hi) * AMAX + nt * 16 + lh] = c[r] + bv;
      }
    }
  }

  // ---- Masked log-softmax, log_prob gather, entropy (lane == action) -----
  for (int rr = 0; rr < ROWS_PER_WAVE; ++rr) {
    const int grow = rowBase + rr;
    const int leaf = leaf_ids[grow];
    const int n    = counts[leaf];
    const bool valid = lane < n;
    float x = valid ? sm.logits[rr * AMAX + lane] : -1e9f;

    float m = x;
#pragma unroll
    for (int off = 16; off; off >>= 1) m = fmaxf(m, __shfl_xor(m, off, 32));
    float e = __builtin_amdgcn_exp2f((x - m) * 1.4426950408889634f);
    float s = e;
#pragma unroll
    for (int off = 16; off; off >>= 1) s += __shfl_xor(s, off, 32);
    float logp = (x - m) - __builtin_amdgcn_logf(s) * 0.6931471805599453f;
    float t = valid ? e * logp : 0.0f;
#pragma unroll
    for (int off = 16; off; off >>= 1) t += __shfl_xor(t, off, 32);

    int act = actions[grow];
    act = act < 0 ? 0 : (act > n - 1 ? n - 1 : act);
    if (lane == act) out_logp[grow] = logp;
    if (lane == 0)   out_ent[grow]  = -t * __builtin_amdgcn_rcpf(s);
  }

  // ---- Value network (x still alive; h reused in place) -------------------
  dense64(v1t, bv1);        // HV1
  dense256(v2t, bv2);       // HV

  {
    float v3r[8];
#pragma unroll
    for (int j = 0; j < 8; ++j) v3r[j] = V3[lane * 8 + j];
    const float bvf = bv3[0];
    for (int rr = 0; rr < ROWS_PER_WAVE; ++rr) {
      H8U u;
      u.q = *(const vi4*)(sm.h + rr * HPAD + lane * 8);
      float acc = 0.0f;
#pragma unroll
      for (int j = 0; j < 8; ++j) acc += (float)u.h[j] * v3r[j];
#pragma unroll
      for (int off = 16; off; off >>= 1) acc += __shfl_xor(acc, off, 32);
      if (lane == 0) out_val[rowBase + rr] = acc + bvf;
    }
  }
}

// ---------------------------------------------------------------------------
// Launch
// ---------------------------------------------------------------------------
extern "C" void kernel_launch(void* const* d_in, const int* in_sizes, int n_in,
                              void* d_out, int out_size, void* d_ws, size_t ws_size,
                              hipStream_t stream) {
  const float* obs      = (const float*)d_in[0];
  const int*   actions  = (const int*)d_in[1];
  const int*   leaf_ids = (const int*)d_in[2];
  const int*   counts   = (const int*)d_in[3];
  const float* W1  = (const float*)d_in[4];
  const float* b1  = (const float*)d_in[5];
  const float* W2  = (const float*)d_in[6];
  const float* b2  = (const float*)d_in[7];
  const float* Wh  = (const float*)d_in[8];
  const float* bh  = (const float*)d_in[9];
  const float* V1  = (const float*)d_in[10];
  const float* bv1 = (const float*)d_in[11];
  const float* V2  = (const float*)d_in[12];
  const float* bv2 = (const float*)d_in[13];
  const float* V3  = (const float*)d_in[14];
  const float* bv3 = (const float*)d_in[15];

  _Float16* ws = (_Float16*)d_ws;
  float* out_logp = (float*)d_out;
  float* out_ent  = out_logp + BATCH;
  float* out_val  = out_logp + 2 * BATCH;

  cac_convert_weights<<<WS_HALVES / 256, 256, 0, stream>>>(W1, W2, Wh, V1, V2, ws);

  cac_fused<<<BATCH / ROWS_PER_BLOCK, 256, 0, stream>>>(
      obs, actions, leaf_ids, counts,
      b1, b2, bh, bv1, bv2, V3, bv3,
      ws, out_logp, out_ent, out_val);
}
